// Attention_decoder_25915832664636
// MI455X (gfx1250) — matrix operand order repaired and checked by reading, not compiled, not run on hardware
//
#include <hip/hip_runtime.h>
#include <hip/hip_bf16.h>

// MI455X / gfx1250: wave32, WMMA 16x16x32 f16->f32. All matmul-shaped work is
// routed through v_wmma_f32_16x16x32_f16 (fp16 operands, fp32 accumulation).
// Each wave computes a 64x64 C tile: 16 chained WMMAs per K=32 step against
// 16 b128 fragment loads (1:1 wmma:vmem), then stages the tile through LDS
// (ds_store_b16 / ds_load_b128) so the global write-out is 16 coalesced
// global_store_b128 per lane instead of 128 scattered b16 stores.

typedef __attribute__((ext_vector_type(16))) _Float16 half16;
typedef __attribute__((ext_vector_type(8)))  _Float16 half8;
typedef __attribute__((ext_vector_type(8)))  float    f32x8;

#define BN_EPS 1e-5f

#define GEMM_STORET 1   // store C transposed: C[n][m] instead of C[m][n]

// ---------------------------------------------------------------------------
// Generic WMMA GEMM:  C[M,N] = A[M,K] * Bt[N,K]^T  (+ bias[n]), fp16 output.
// A row-major [M][K] fp16, Bt row-major [N][K] fp16 (B transposed) so both
// operand fragments are contiguous-K 16B loads matching the ISA lane map:
//   A frag (16x32): lane l holds row m0+(l&15), K chunks {kc..kc+7, kc+16..kc+23},
//                   kc = 8 for lanes 16..31.
//   B frag (32x16): lane l holds col n0+(l&15), K chunks {kc..kc+7, kc+16..kc+23}, kc = 8 for lanes>=16.
//   C frag (16x16): VGPR r, lane l -> (m = r + 8*(l>=16), n = l&15).
// One wave (32 threads) per block, 64x64 tile = 4x4 subtiles.
// All M,N used here are multiples of 64, K multiples of 32 -> no bounds checks,
// EXEC all-ones (WMMA requirement).
// ---------------------------------------------------------------------------
__global__ __launch_bounds__(32)
void gemm_wmma_kernel(const _Float16* __restrict__ A,
                      const _Float16* __restrict__ Bt,
                      _Float16* __restrict__ Cout,
                      const float* __restrict__ bias,
                      int K, int lda, int ldb, int ldc,
                      long sA, long sB, long sC, int flags)
{
  __shared__ _Float16 lsc[64 * 72];   // 64x64 tile, stride 72 (16B-aligned rows)

  const int lane = threadIdx.x;
  const int n0 = blockIdx.x * 64;
  const int m0 = blockIdx.y * 64;
  const int b  = blockIdx.z;
  A  += (long)b * sA;
  Bt += (long)b * sB;

  const int l15 = lane & 15;
  const int kcA = (lane >> 4) << 3;   // 0 or 8
  const int kcB = (lane >> 4) << 3;   // 0 or 8
  const _Float16* arow = A + (long)(m0 + l15) * lda + kcA;
  const _Float16* brow = Bt + (long)(n0 + l15) * ldb + kcB;

  f32x8 acc[4][4];
#pragma unroll
  for (int i = 0; i < 4; ++i)
#pragma unroll
    for (int j = 0; j < 4; ++j)
      acc[i][j] = (f32x8){};

  for (int k0 = 0; k0 < K; k0 += 32) {
    half16 af[4], bf[4];
#pragma unroll
    for (int i = 0; i < 4; ++i) {
      const _Float16* pa = arow + (long)(16 * i) * lda + k0;
      half8 lo = *(const half8*)(pa);
      half8 hi = *(const half8*)(pa + 16);
      af[i] = __builtin_shufflevector(lo, hi, 0,1,2,3,4,5,6,7,8,9,10,11,12,13,14,15);
    }
#pragma unroll
    for (int j = 0; j < 4; ++j) {
      const _Float16* pb = brow + (long)(16 * j) * ldb + k0;
      half8 lo = *(const half8*)(pb);
      half8 hi = *(const half8*)(pb + 16);
      bf[j] = __builtin_shufflevector(lo, hi, 0,1,2,3,4,5,6,7,8,9,10,11,12,13,14,15);
    }

    if (k0 + 32 < K) {  // gfx1250 global_prefetch_b8 of next K block
#pragma unroll
      for (int i = 0; i < 4; ++i)
        __builtin_prefetch(arow + (long)(16 * i) * lda + k0 + 32, 0, 3);
#pragma unroll
      for (int j = 0; j < 4; ++j)
        __builtin_prefetch(brow + (long)(16 * j) * ldb + k0 + 32, 0, 3);
    }

#pragma unroll
    for (int i = 0; i < 4; ++i)
#pragma unroll
      for (int j = 0; j < 4; ++j)
        acc[i][j] = __builtin_amdgcn_wmma_f32_16x16x32_f16(false, af[i], false, bf[j],
                                                           (short)0, acc[i][j], false, false);
  }

  // ---- epilogue: bias add + LDS transpose stage, then coalesced b128 stores ----
  float bn[4];
#pragma unroll
  for (int j = 0; j < 4; ++j)
    bn[j] = bias ? bias[n0 + 16 * j + l15] : 0.f;

  const bool storet = (flags & GEMM_STORET) != 0;
  const int mloc = (lane >> 4) << 3;        // local m base: 0 or 8
#pragma unroll
  for (int i = 0; i < 4; ++i) {
#pragma unroll
    for (int j = 0; j < 4; ++j) {
      const int nL = 16 * j + l15;
#pragma unroll
      for (int r = 0; r < 8; ++r) {
        const int mL = mloc + 16 * i + r;
        float v = acc[i][j][r] + bn[j];
        const int a = storet ? (nL * 72 + mL) : (mL * 72 + nL);
        lsc[a] = (_Float16)v;
      }
    }
  }
  __syncthreads();  // single-wave WG: orders LDS writes before reads

  const long row0 = storet ? n0 : m0;
  const long col0 = storet ? m0 : n0;
  _Float16* Cp = Cout + (long)b * sC + row0 * ldc + col0;
#pragma unroll
  for (int it = 0; it < 16; ++it) {
    const int rr = it * 4 + (lane >> 3);    // 4 rows per iteration
    const int cc = (lane & 7) * 8;          // 8 halves (16B) per lane
    half8 vv = *(const half8*)&lsc[rr * 72 + cc];
    *(half8*)(Cp + (long)rr * ldc + cc) = vv;
  }
}

// ---------------------------------------------------------------------------
// Prep kernels: fold eval-mode BN into fp16 weights + fp32 bias.
// y = (W x + b)*s + (bb - m*s),  s = g/sqrt(v+eps)  ->  W' = W*s, b' = b*s + bb - m*s
// ---------------------------------------------------------------------------
__global__ void fold_pw_kernel(const float* __restrict__ w, const float* __restrict__ bin,
                               const float* __restrict__ g, const float* __restrict__ bb,
                               const float* __restrict__ mean, const float* __restrict__ var,
                               float extra, _Float16* __restrict__ wh, float* __restrict__ bout,
                               int O, int C)
{
  int idx = blockIdx.x * blockDim.x + threadIdx.x;
  if (idx >= O * C) return;
  int o = idx / C;
  float s = 1.f;
  if (g) s = g[o] * rsqrtf(var[o] + BN_EPS);
  wh[idx] = (_Float16)(w[idx] * s * extra);
  if ((idx % C) == 0) {
    float bi = bin ? bin[o] : 0.f;
    float sh = g ? (bb[o] - mean[o] * s) : 0.f;
    bout[o] = bi * s + sh;
  }
}

// conv_w (O,I,3,3) + bn1  ->  Wc[o][tap*256 + i] fp16, tap = kh*3+kw
__global__ void prep_conv_kernel(const float* __restrict__ w, const float* __restrict__ bin,
                                 const float* __restrict__ g, const float* __restrict__ bb,
                                 const float* __restrict__ mean, const float* __restrict__ var,
                                 _Float16* __restrict__ wh, float* __restrict__ bout)
{
  int idx = blockIdx.x * blockDim.x + threadIdx.x;
  if (idx >= 256 * 2304) return;
  int o = idx / 2304;
  int r = idx % 2304;
  int tap = r >> 8;
  int i = r & 255;
  float s = g[o] * rsqrtf(var[o] + BN_EPS);
  wh[idx] = (_Float16)(w[(o * 256 + i) * 9 + tap] * s);
  if (r == 0) bout[o] = bin[o] * s + bb[o] - mean[o] * s;
}

// tr_w (I,O,3,3) -> Wtr[(tap*256 + o)][i] fp16   (direct convT formula: oh = 2h+kh-1)
__global__ void prep_tr_kernel(const float* __restrict__ w, _Float16* __restrict__ wh)
{
  int idx = blockIdx.x * blockDim.x + threadIdx.x;
  if (idx >= 2304 * 256) return;
  int i = idx & 255;
  int row = idx >> 8;       // tap*256 + o
  int o = row & 255;
  int tap = row >> 8;
  wh[idx] = (_Float16)(w[(i * 256 + o) * 9 + tap]);
}

// x (B,256,64,64) fp32 NCHW -> im2col fp16 [B*4096][2304], K = tap*256 + i, pad=1
__global__ void im2col_kernel(const float* __restrict__ x, _Float16* __restrict__ col)
{
  long idx = (long)blockIdx.x * blockDim.x + threadIdx.x;
  if (idx >= (long)16384 * 2304) return;
  int kk = (int)(idx % 2304);
  long row = idx / 2304;
  int tap = kk >> 8, i = kk & 255;
  int wc = (int)(row & 63);
  int h  = (int)((row >> 6) & 63);
  int b  = (int)(row >> 12);
  int hh = h + (tap / 3) - 1;
  int ww = wc + (tap % 3) - 1;
  float v = 0.f;
  if (hh >= 0 && hh < 64 && ww >= 0 && ww < 64)
    v = x[(((long)b * 256 + i) << 12) + (hh << 6) + ww];
  col[idx] = (_Float16)v;
}

// x_conv fp32 NCHW -> fp16 into xcat[:, 256:512] (ldc = 512)
__global__ void xconv_to_xcat_kernel(const float* __restrict__ xc, _Float16* __restrict__ xcat)
{
  int idx = blockIdx.x * blockDim.x + threadIdx.x;
  if (idx >= 16384 * 256) return;
  int c = idx & 255;
  int row = idx >> 8;      // b*4096 + s
  int s = row & 4095;
  int b = row >> 12;
  xcat[(long)row * 512 + 256 + c] = (_Float16)xc[(((long)b * 256 + c) << 12) + s];
}

// ConvTranspose2d(3, s=2, p=1, op=1) gather: out[b,o,oh,ow] = trb[o] +
//   sum over taps with oh = 2h+kh-1, ow = 2w+kw-1 of taps[b*4096+h*64+w][tap*256+o]
__global__ void convt_assemble_kernel(const _Float16* __restrict__ taps,
                                      const float* __restrict__ trb,
                                      float* __restrict__ out)
{
  long idx = (long)blockIdx.x * blockDim.x + threadIdx.x;
  if (idx >= (long)4 * 256 * 128 * 128) return;
  int ow = (int)(idx & 127);
  int oh = (int)((idx >> 7) & 127);
  int o  = (int)((idx >> 14) & 255);
  int b  = (int)(idx >> 22);
  float acc = trb[o];
#pragma unroll
  for (int kh = 0; kh < 3; ++kh) {
    int hh = oh + 1 - kh;
    if (hh & 1) continue;
    int h = hh >> 1;
    if (h < 0 || h >= 64) continue;
#pragma unroll
    for (int kw = 0; kw < 3; ++kw) {
      int wn = ow + 1 - kw;
      if (wn & 1) continue;
      int w = wn >> 1;
      if (w < 0 || w >= 64) continue;
      long row = ((long)b << 12) + (h << 6) + w;
      acc += (float)taps[row * 2304 + (long)((kh * 3 + kw) * 256 + o)];
    }
  }
  out[idx] = acc;
}

// ---------------------------------------------------------------------------
// Workspace layout (bytes). BIG region is time-shared: im2col -> mulT -> taps.
// ---------------------------------------------------------------------------
static constexpr size_t AL(size_t x) { return (x + 255) & ~(size_t)255; }
static constexpr size_t O_WC   = 0;
static constexpr size_t O_WQ   = AL(O_WC   + (size_t)256 * 2304 * 2);
static constexpr size_t O_WK   = AL(O_WQ   + (size_t)128 * 256 * 2);
static constexpr size_t O_WV   = AL(O_WK   + (size_t)128 * 256 * 2);
static constexpr size_t O_WOUT = AL(O_WV   + (size_t)128 * 256 * 2);
static constexpr size_t O_WO1  = AL(O_WOUT + (size_t)256 * 128 * 2);
static constexpr size_t O_WTR  = AL(O_WO1  + (size_t)256 * 512 * 2);
static constexpr size_t O_BC   = AL(O_WTR  + (size_t)2304 * 256 * 2);
static constexpr size_t O_BQ   = AL(O_BC   + 256 * 4);
static constexpr size_t O_BK   = AL(O_BQ   + 128 * 4);
static constexpr size_t O_BV   = AL(O_BK   + 128 * 4);
static constexpr size_t O_BOUT = AL(O_BV   + 128 * 4);
static constexpr size_t O_BO1  = AL(O_BOUT + 256 * 4);
static constexpr size_t O_X    = AL(O_BO1  + 256 * 4);
static constexpr size_t O_QH   = AL(O_X    + (size_t)16384 * 256 * 2);
static constexpr size_t O_KH   = AL(O_QH   + (size_t)16384 * 128 * 2);
static constexpr size_t O_VT   = AL(O_KH   + (size_t)16384 * 128 * 2);
static constexpr size_t O_ATT  = AL(O_VT   + (size_t)16384 * 128 * 2);
static constexpr size_t O_XCAT = AL(O_ATT  + (size_t)16384 * 128 * 2);
static constexpr size_t O_XO   = AL(O_XCAT + (size_t)16384 * 512 * 2);
static constexpr size_t O_BIG  = AL(O_XO   + (size_t)16384 * 256 * 2);
static constexpr size_t WS_NEED = O_BIG + (size_t)4 * 4096 * 4096 * 2;

static inline void gemm(const void* A, const void* Bt, void* C, const void* bias,
                        int M, int N, int K, int lda, int ldb, int ldc,
                        long sA, long sB, long sC, int nb, int flags, hipStream_t st)
{
  dim3 grid((unsigned)(N / 64), (unsigned)(M / 64), (unsigned)nb);
  gemm_wmma_kernel<<<grid, dim3(32), 0, st>>>((const _Float16*)A, (const _Float16*)Bt,
                                              (_Float16*)C, (const float*)bias,
                                              K, lda, ldb, ldc, sA, sB, sC, flags);
}

extern "C" void kernel_launch(void* const* d_in, const int* in_sizes, int n_in,
                              void* d_out, int out_size, void* d_ws, size_t ws_size,
                              hipStream_t stream)
{
  (void)in_sizes; (void)n_in; (void)out_size;
  const float* x      = (const float*)d_in[0];
  const float* x_conv = (const float*)d_in[1];
  const float* conv_w = (const float*)d_in[2];
  const float* conv_b = (const float*)d_in[3];
  const float* bn1_g  = (const float*)d_in[4];
  const float* bn1_b  = (const float*)d_in[5];
  const float* bn1_m  = (const float*)d_in[6];
  const float* bn1_v  = (const float*)d_in[7];
  const float* q_w    = (const float*)d_in[8];
  const float* q_b    = (const float*)d_in[9];
  const float* qbn_g  = (const float*)d_in[10];
  const float* qbn_b  = (const float*)d_in[11];
  const float* qbn_m  = (const float*)d_in[12];
  const float* qbn_v  = (const float*)d_in[13];
  const float* k_w    = (const float*)d_in[14];
  const float* k_b    = (const float*)d_in[15];
  const float* kbn_g  = (const float*)d_in[16];
  const float* kbn_b  = (const float*)d_in[17];
  const float* kbn_m  = (const float*)d_in[18];
  const float* kbn_v  = (const float*)d_in[19];
  const float* v_w    = (const float*)d_in[20];
  const float* v_b    = (const float*)d_in[21];
  const float* out_w  = (const float*)d_in[22];
  const float* out_b  = (const float*)d_in[23];
  const float* out1_w = (const float*)d_in[24];
  const float* out1_b = (const float*)d_in[25];
  const float* obn_g  = (const float*)d_in[26];
  const float* obn_b  = (const float*)d_in[27];
  const float* obn_m  = (const float*)d_in[28];
  const float* obn_v  = (const float*)d_in[29];
  const float* tr_w   = (const float*)d_in[30];
  const float* tr_b   = (const float*)d_in[31];

  if (ws_size < WS_NEED) return;
  char* ws = (char*)d_ws;
  _Float16* WC   = (_Float16*)(ws + O_WC);
  _Float16* WQ   = (_Float16*)(ws + O_WQ);
  _Float16* WK   = (_Float16*)(ws + O_WK);
  _Float16* WV   = (_Float16*)(ws + O_WV);
  _Float16* WOUT = (_Float16*)(ws + O_WOUT);
  _Float16* WO1  = (_Float16*)(ws + O_WO1);
  _Float16* WTR  = (_Float16*)(ws + O_WTR);
  float* BC   = (float*)(ws + O_BC);
  float* BQ   = (float*)(ws + O_BQ);
  float* BK   = (float*)(ws + O_BK);
  float* BV   = (float*)(ws + O_BV);
  float* BOUT = (float*)(ws + O_BOUT);
  float* BO1  = (float*)(ws + O_BO1);
  _Float16* Xh   = (_Float16*)(ws + O_X);
  _Float16* QH   = (_Float16*)(ws + O_QH);
  _Float16* KHp  = (_Float16*)(ws + O_KH);
  _Float16* VT   = (_Float16*)(ws + O_VT);
  _Float16* ATT  = (_Float16*)(ws + O_ATT);
  _Float16* XCAT = (_Float16*)(ws + O_XCAT);
  _Float16* XO   = (_Float16*)(ws + O_XO);
  _Float16* COL  = (_Float16*)(ws + O_BIG);   // lifetime: im2col -> conv3 GEMM
  _Float16* MULT = (_Float16*)(ws + O_BIG);   // lifetime: mulT GEMM -> att GEMM
  _Float16* TAPS = (_Float16*)(ws + O_BIG);   // lifetime: tap GEMM -> assemble

  const int T = 256;
  // ---- weight prep (BN folding, reordering, 1/sqrt(256) into out_w) ----
  fold_pw_kernel<<<(128 * 256 + T - 1) / T, T, 0, stream>>>(q_w, q_b, qbn_g, qbn_b, qbn_m, qbn_v, 1.f, WQ, BQ, 128, 256);
  fold_pw_kernel<<<(128 * 256 + T - 1) / T, T, 0, stream>>>(k_w, k_b, kbn_g, kbn_b, kbn_m, kbn_v, 1.f, WK, BK, 128, 256);
  fold_pw_kernel<<<(128 * 256 + T - 1) / T, T, 0, stream>>>(v_w, v_b, nullptr, nullptr, nullptr, nullptr, 1.f, WV, BV, 128, 256);
  fold_pw_kernel<<<(256 * 128 + T - 1) / T, T, 0, stream>>>(out_w, out_b, nullptr, nullptr, nullptr, nullptr, 1.f / 16.f, WOUT, BOUT, 256, 128);
  fold_pw_kernel<<<(256 * 512 + T - 1) / T, T, 0, stream>>>(out1_w, out1_b, obn_g, obn_b, obn_m, obn_v, 1.f, WO1, BO1, 256, 512);
  prep_conv_kernel<<<(256 * 2304 + T - 1) / T, T, 0, stream>>>(conv_w, conv_b, bn1_g, bn1_b, bn1_m, bn1_v, WC, BC);
  prep_tr_kernel<<<(2304 * 256 + T - 1) / T, T, 0, stream>>>(tr_w, WTR);

  // ---- conv3x3 + BN1 as implicit GEMM (im2col, K=2304) -> X fp16 NHWC ----
  im2col_kernel<<<(int)(((long)16384 * 2304 + T - 1) / T), T, 0, stream>>>(x, COL);
  gemm(COL, WC, Xh, BC, 16384, 256, 2304, 2304, 2304, 256, 0, 0, 0, 1, 0, stream);

  // ---- q, k (BN folded); v stored transposed per batch: VT[b][c][s] ----
  gemm(Xh, WQ, QH, BQ, 16384, 128, 256, 256, 256, 128, 0, 0, 0, 1, 0, stream);
  gemm(Xh, WK, KHp, BK, 16384, 128, 256, 256, 256, 128, 0, 0, 0, 1, 0, stream);
  gemm(Xh, WV, VT, BV, 4096, 128, 256, 256, 256, 4096,
       (long)4096 * 256, 0, (long)128 * 4096, 4, GEMM_STORET, stream);

  // ---- attention (no softmax): mulT[m][n] = sum_c K[m,c]*Q[n,c] ----
  gemm(KHp, QH, MULT, nullptr, 4096, 4096, 128, 128, 128, 4096,
       (long)4096 * 128, (long)4096 * 128, (long)4096 * 4096, 4, 0, stream);
  // att[m][c] = sum_n mulT[m][n] * VT[c][n]
  gemm(MULT, VT, ATT, nullptr, 4096, 128, 4096, 4096, 4096, 128,
       (long)4096 * 4096, (long)128 * 4096, (long)4096 * 128, 4, 0, stream);

  // ---- out-proj (with /16 folded) -> xcat[:,0:256]; x_conv -> xcat[:,256:512] ----
  gemm(ATT, WOUT, XCAT, BOUT, 16384, 256, 128, 128, 128, 512, 0, 0, 0, 1, 0, stream);
  xconv_to_xcat_kernel<<<(16384 * 256 + T - 1) / T, T, 0, stream>>>(x_conv, XCAT);

  // ---- out1 + obn (folded), K=512 ----
  gemm(XCAT, WO1, XO, BO1, 16384, 256, 512, 512, 512, 256, 0, 0, 0, 1, 0, stream);

  // ---- transposed conv as 9-tap GEMM (N=2304) + parity-gather assemble ----
  gemm(XO, WTR, TAPS, nullptr, 16384, 2304, 256, 256, 256, 2304, 0, 0, 0, 1, 0, stream);
  convt_assemble_kernel<<<(int)(((long)4 * 256 * 128 * 128 + T - 1) / T), T, 0, stream>>>(TAPS, tr_b, (float*)d_out);
}
